// CryptoAttentionLayer_29317446762898
// MI455X (gfx1250) — matrix operand order repaired and checked by reading, not compiled, not run on hardware
//
#include <hip/hip_runtime.h>
#include <hip/hip_bf16.h>

// ---------- types for WMMA ----------
typedef __attribute__((ext_vector_type(16))) __bf16 v16bf;
typedef __attribute__((ext_vector_type(8)))  float  v8f;
typedef __attribute__((ext_vector_type(4)))  int    v4i;

union FragBF { uint4 u[2]; v16bf v; };

// ---------- optional gfx1250 async global->LDS copy ----------
#if __has_builtin(__builtin_amdgcn_global_load_async_to_lds_b128) && \
    __has_builtin(__builtin_amdgcn_s_wait_asynccnt)
#define HAVE_ASYNC_LDS 1
#else
#define HAVE_ASYNC_LDS 0
#endif

#if HAVE_ASYNC_LDS
typedef __attribute__((address_space(1))) v4i gv4i;   // global int4
typedef __attribute__((address_space(3))) v4i lv4i;   // LDS int4

__device__ __forceinline__ void async_copy_b128(const void* g, void* l) {
    __builtin_amdgcn_global_load_async_to_lds_b128(
        (gv4i*)(unsigned long long)g,
        (lv4i*)(unsigned long long)l,
        0, 0);
}
__device__ __forceinline__ void async_drain() {
    __builtin_amdgcn_s_wait_asynccnt(0);
}
#endif

// ---------- bf16 helpers ----------
__device__ __forceinline__ float bf2f(unsigned int b) {
    return __uint_as_float(b << 16);
}
__device__ __forceinline__ unsigned short f2bf(float f) {
    unsigned int u = __float_as_uint(f);
    unsigned int r = ((u >> 16) & 1u) + 0x7FFFu;   // round-to-nearest-even
    return (unsigned short)((u + r) >> 16);
}
__device__ __forceinline__ unsigned short u4get(const uint4& v, int j) {
    unsigned w = (j < 2) ? v.x : (j < 4) ? v.y : (j < 6) ? v.z : v.w;
    return (unsigned short)((j & 1) ? (w >> 16) : (w & 0xFFFFu));
}
__device__ __forceinline__ void unpack8(const uint4& u, float* f) {
    f[0] = bf2f(u.x & 0xFFFFu); f[1] = bf2f(u.x >> 16);
    f[2] = bf2f(u.y & 0xFFFFu); f[3] = bf2f(u.y >> 16);
    f[4] = bf2f(u.z & 0xFFFFu); f[5] = bf2f(u.z >> 16);
    f[6] = bf2f(u.w & 0xFFFFu); f[7] = bf2f(u.w >> 16);
}
__device__ __forceinline__ uint4 pack8(const float* f) {
    uint4 u;
    u.x = (unsigned)f2bf(f[0]) | ((unsigned)f2bf(f[1]) << 16);
    u.y = (unsigned)f2bf(f[2]) | ((unsigned)f2bf(f[3]) << 16);
    u.z = (unsigned)f2bf(f[4]) | ((unsigned)f2bf(f[5]) << 16);
    u.w = (unsigned)f2bf(f[6]) | ((unsigned)f2bf(f[7]) << 16);
    return u;
}
__device__ __forceinline__ v8f wmma_bf16(const FragBF& a, const FragBF& b, v8f c) {
    return __builtin_amdgcn_wmma_f32_16x16x32_bf16(
        false, a.v, false, b.v, (short)0, c, false, false);
}

// ---------- f32 -> bf16 conversion (vectorized) ----------
__global__ __launch_bounds__(256)
void k_f32_to_bf16(const float* __restrict__ src, unsigned short* __restrict__ dst, int n4) {
    int i = blockIdx.x * blockDim.x + threadIdx.x;
    if (i >= n4) return;
    float4 v = ((const float4*)src)[i];
    uint2 p;
    p.x = (unsigned)f2bf(v.x) | ((unsigned)f2bf(v.y) << 16);
    p.y = (unsigned)f2bf(v.z) | ((unsigned)f2bf(v.w) << 16);
    ((uint2*)dst)[i] = p;
}

// ---------- tiled bf16 WMMA GEMM: C = A(MxK) @ B(KxN) + bias ----------
// Block tile 64(M) x 128(N), BK=32, double-buffered LDS, 256 threads = 8 waves
// in a 2x4 grid. Each wave computes a 32x32 C tile = 2x2 WMMA tiles -> 4
// v_wmma_f32_16x16x32_bf16 per K-step with A/B fragment reuse.
// A tile is staged with GLOBAL_LOAD_ASYNC_TO_LDS_B128 when available
// (ASYNCcnt-tracked DMA, drained before the barrier); B tile is transposed by
// VALU into [N][K] so B fragments are per-lane K-contiguous b128 reads.
template<bool OUT_BF16>
__global__ __launch_bounds__(256)
void k_gemm_bias(const unsigned short* __restrict__ A,
                 const unsigned short* __restrict__ Bm,
                 const float* __restrict__ bias,
                 void* __restrict__ C,
                 int K, int N) {
    __shared__ __align__(16) unsigned short As[2][64][40];
    __shared__ __align__(16) unsigned short Bt[2][128][40];

    const int t    = threadIdx.x;
    const int lane = t & 31;
    const int wave = t >> 5;
    const int mw   = wave >> 2;          // 0..1 : 32-row group
    const int nw   = wave & 3;           // 0..3 : 32-col group
    const int rowBase = blockIdx.x * 64;
    const int colBase = blockIdx.y * 128;

    // A staging: 64 rows x 32 cols, one uint4 (8 bf16) per thread
    const int ar = t >> 2;               // 0..63
    const int ac = (t & 3) * 8;          // 0,8,16,24
    // B staging: 16 K-row-pairs x 128 cols, 8 cols per thread, pack K-pairs
    const int bp  = t >> 4;              // 0..15  -> K rows 2bp, 2bp+1
    const int bcL = (t & 15) * 8;        // 0..120

    const int fr = lane & 15;            // fragment row/col within 16
    const int fh = lane >> 4;            // K-half select

    const unsigned short* Aptr  = A  + (size_t)(rowBase + ar) * K + ac;
    const unsigned short* Bptr0 = Bm + (size_t)(2 * bp) * N + colBase + bcL;
    const unsigned short* Bptr1 = Bptr0 + N;

    v8f acc00 = {}, acc01 = {}, acc10 = {}, acc11 = {};

    const int nsteps = K / 32;

    // ---- prologue: stage tile 0 into buffer 0 ----
    {
#if HAVE_ASYNC_LDS
        async_copy_b128(Aptr, &As[0][ar][ac]);
#else
        uint4 a0v = *(const uint4*)Aptr;
        *(uint4*)&As[0][ar][ac] = a0v;
#endif
        uint4 b0v = *(const uint4*)Bptr0;
        uint4 b1v = *(const uint4*)Bptr1;
#pragma unroll
        for (int j = 0; j < 8; ++j) {
            unsigned p = (unsigned)u4get(b0v, j) | ((unsigned)u4get(b1v, j) << 16);
            *(unsigned*)&Bt[0][bcL + j][2 * bp] = p;
        }
#if HAVE_ASYNC_LDS
        async_drain();
#endif
    }
    __syncthreads();

    uint4 bv0, bv1;
    for (int s = 0; s < nsteps; ++s) {
        const int cur = s & 1;
        const int nxt = cur ^ 1;
        const bool more = (s + 1 < nsteps);

        // ---- issue loads for step s+1 (latency hidden under the WMMAs) ----
#if !HAVE_ASYNC_LDS
        uint4 aval;
#endif
        if (more) {
            const int ko = (s + 1) * 32;
#if HAVE_ASYNC_LDS
            async_copy_b128(Aptr + ko, &As[nxt][ar][ac]);   // DMA, ASYNCcnt
#else
            aval = *(const uint4*)(Aptr + ko);
#endif
            bv0 = *(const uint4*)(Bptr0 + (size_t)ko * N);
            bv1 = *(const uint4*)(Bptr1 + (size_t)ko * N);
        }
        if (s + 2 < nsteps) {   // gfx1250 global_prefetch_b8 two steps ahead
            const int kp = (s + 2) * 32;
            __builtin_prefetch(Aptr + kp, 0, 1);
            __builtin_prefetch(Bptr0 + (size_t)kp * N, 0, 1);
        }

        // ---- fragments from current buffer ----
        FragBF a0, a1, b0, b1;
        a0.u[0] = *(const uint4*)&As[cur][mw * 32 + fr     ][fh * 8];
        a0.u[1] = *(const uint4*)&As[cur][mw * 32 + fr     ][16 + fh * 8];
        a1.u[0] = *(const uint4*)&As[cur][mw * 32 + 16 + fr][fh * 8];
        a1.u[1] = *(const uint4*)&As[cur][mw * 32 + 16 + fr][16 + fh * 8];
        b0.u[0] = *(const uint4*)&Bt[cur][nw * 32 + fr     ][fh * 8];
        b0.u[1] = *(const uint4*)&Bt[cur][nw * 32 + fr     ][16 + fh * 8];
        b1.u[0] = *(const uint4*)&Bt[cur][nw * 32 + 16 + fr][fh * 8];
        b1.u[1] = *(const uint4*)&Bt[cur][nw * 32 + 16 + fr][16 + fh * 8];

        acc00 = wmma_bf16(a0, b0, acc00);
        acc01 = wmma_bf16(a0, b1, acc01);
        acc10 = wmma_bf16(a1, b0, acc10);
        acc11 = wmma_bf16(a1, b1, acc11);

        // ---- stage step s+1 into the other buffer ----
        if (more) {
#if !HAVE_ASYNC_LDS
            *(uint4*)&As[nxt][ar][ac] = aval;
#endif
#pragma unroll
            for (int j = 0; j < 8; ++j) {
                unsigned p = (unsigned)u4get(bv0, j) | ((unsigned)u4get(bv1, j) << 16);
                *(unsigned*)&Bt[nxt][bcL + j][2 * bp] = p;
            }
#if HAVE_ASYNC_LDS
            async_drain();   // ASYNCcnt==0 before the barrier
#endif
        }
        __syncthreads();
    }

    // ---- epilogue: bias + store ----
    // C/D layout: lane%16 = column, VGPR v = row v (lanes 0-15) / v+8 (16-31)
    const int rsel = (lane < 16) ? 0 : 8;
#pragma unroll
    for (int tj = 0; tj < 2; ++tj) {
        const int col = colBase + nw * 32 + tj * 16 + fr;
        const float bc_ = bias ? bias[col] : 0.0f;
#pragma unroll
        for (int ti = 0; ti < 2; ++ti) {
            const v8f& acc = (ti == 0) ? (tj == 0 ? acc00 : acc01)
                                       : (tj == 0 ? acc10 : acc11);
#pragma unroll
            for (int vI = 0; vI < 8; ++vI) {
                const int row = rowBase + mw * 32 + ti * 16 + vI + rsel;
                const float val = acc[vI] + bc_;
                if (OUT_BF16) {
                    ((unsigned short*)C)[(size_t)row * N + col] = f2bf(val);
                } else {
                    // write-once 128MB result: keep it out of the caches
                    __builtin_nontemporal_store(
                        val, &((float*)C)[(size_t)row * N + col]);
                }
            }
        }
    }
}

// ---------- per-node head-head attention (wave32 per node) ----------
// scores[h][g] = <Q[h,:], K[g,:]> / 16 ; softmax over g ; out = W @ V
__global__ __launch_bounds__(256)
void k_node_head_attn(const unsigned short* __restrict__ Q,
                      const unsigned short* __restrict__ Kq,
                      const unsigned short* __restrict__ V,
                      unsigned short* __restrict__ O) {
    const int lane = threadIdx.x & 31;
    const int node = blockIdx.x * 8 + (threadIdx.x >> 5);
    const size_t base = (size_t)node * 1024 + (size_t)lane * 8;

    float qf[4][8], kf[4][8], vf[4][8];
#pragma unroll
    for (int h = 0; h < 4; ++h) {
        uint4 uq = *(const uint4*)&Q [base + h * 256];
        uint4 uk = *(const uint4*)&Kq[base + h * 256];
        uint4 uv = *(const uint4*)&V [base + h * 256];
        unpack8(uq, qf[h]); unpack8(uk, kf[h]); unpack8(uv, vf[h]);
    }

    float s[4][4];
#pragma unroll
    for (int h = 0; h < 4; ++h) {
#pragma unroll
        for (int g = 0; g < 4; ++g) {
            float p = 0.0f;
#pragma unroll
            for (int j = 0; j < 8; ++j) p += qf[h][j] * kf[g][j];
#pragma unroll
            for (int off = 16; off > 0; off >>= 1) p += __shfl_xor(p, off, 32);
            s[h][g] = p * 0.0625f;   // 1/sqrt(256)
        }
    }

    float w[4][4];
#pragma unroll
    for (int h = 0; h < 4; ++h) {
        float m = fmaxf(fmaxf(s[h][0], s[h][1]), fmaxf(s[h][2], s[h][3]));
        float e0 = __expf(s[h][0] - m), e1 = __expf(s[h][1] - m);
        float e2 = __expf(s[h][2] - m), e3 = __expf(s[h][3] - m);
        float inv = 1.0f / (e0 + e1 + e2 + e3);
        w[h][0] = e0 * inv; w[h][1] = e1 * inv; w[h][2] = e2 * inv; w[h][3] = e3 * inv;
    }

#pragma unroll
    for (int h = 0; h < 4; ++h) {
        float ov[8];
#pragma unroll
        for (int j = 0; j < 8; ++j)
            ov[j] = w[h][0] * vf[0][j] + w[h][1] * vf[1][j]
                  + w[h][2] * vf[2][j] + w[h][3] * vf[3][j];
        *(uint4*)&O[base + h * 256] = pack8(ov);
    }
}

// ---------- host launch ----------
extern "C" void kernel_launch(void* const* d_in, const int* in_sizes, int n_in,
                              void* d_out, int out_size, void* d_ws, size_t ws_size,
                              hipStream_t stream) {
    const float* x   = (const float*)d_in[0];
    // d_in[1] = edge_index (unused by the reference forward)
    const float* Wq  = (const float*)d_in[2];
    const float* bq  = (const float*)d_in[3];
    const float* Wk  = (const float*)d_in[4];
    const float* bk  = (const float*)d_in[5];
    const float* Wv  = (const float*)d_in[6];
    const float* bvv = (const float*)d_in[7];
    const float* Wo  = (const float*)d_in[8];
    const float* bo  = (const float*)d_in[9];
    float* out = (float*)d_out;

    const int Mrows = 32768;   // B*N = 8*4096
    const int Kdim  = 1024;
    const int Ndim  = 1024;

    char* ws = (char*)d_ws;
    const size_t MB = 1024u * 1024u;
    unsigned short* xb  = (unsigned short*)(ws);              // 64 MB  (x in bf16)
    unsigned short* wqb = (unsigned short*)(ws +  64 * MB);   //  2 MB
    unsigned short* wkb = (unsigned short*)(ws +  66 * MB);
    unsigned short* wvb = (unsigned short*)(ws +  68 * MB);
    unsigned short* wob = (unsigned short*)(ws +  70 * MB);
    unsigned short* Qb  = (unsigned short*)(ws +  72 * MB);   // 64 MB
    unsigned short* Kb  = (unsigned short*)(ws + 136 * MB);   // 64 MB
    unsigned short* Vb  = (unsigned short*)(ws + 200 * MB);   // 64 MB
    unsigned short* Ab  = xb;  // attended output reuses the xb region

    // 1) precision conversion
    k_f32_to_bf16<<<32768, 256, 0, stream>>>(x,  xb,  (Mrows * Kdim) / 4);
    k_f32_to_bf16<<<1024,  256, 0, stream>>>(Wq, wqb, (Kdim * Ndim) / 4);
    k_f32_to_bf16<<<1024,  256, 0, stream>>>(Wk, wkb, (Kdim * Ndim) / 4);
    k_f32_to_bf16<<<1024,  256, 0, stream>>>(Wv, wvb, (Kdim * Ndim) / 4);
    k_f32_to_bf16<<<1024,  256, 0, stream>>>(Wo, wob, (Kdim * Ndim) / 4);

    // 2) Q/K/V projections (bf16 WMMA, bf16 out)
    dim3 g(Mrows / 64, Ndim / 128);
    k_gemm_bias<true ><<<g, 256, 0, stream>>>(xb, wqb, bq,  Qb, Kdim, Ndim);
    k_gemm_bias<true ><<<g, 256, 0, stream>>>(xb, wkb, bk,  Kb, Kdim, Ndim);
    k_gemm_bias<true ><<<g, 256, 0, stream>>>(xb, wvb, bvv, Vb, Kdim, Ndim);

    // 3) per-node head attention
    k_node_head_attn<<<Mrows / 8, 256, 0, stream>>>(Qb, Kb, Vb, Ab);

    // 4) output projection (f32 out + bias)
    k_gemm_bias<false><<<g, 256, 0, stream>>>(Ab, wob, bo, out, Kdim, Ndim);
}
